// Segmentation_52493090291814
// MI455X (gfx1250) — matrix operand order repaired and checked by reading, not compiled, not run on hardware
//
#include <hip/hip_runtime.h>

typedef __attribute__((ext_vector_type(16))) __bf16 v16bf;
typedef __attribute__((ext_vector_type(8)))  __bf16 v8bf;
typedef __attribute__((ext_vector_type(8)))  float  v8f;

#define S_LEN   2048
#define DMODEL  512
#define DFF     2048
#define NHEAD   8
#define DHEAD   64
#define NLAYERS 6

// ---------- helpers ----------
// Hardware packed convert: src0 -> [15:0], src1 -> [31:16], RNE.
__device__ __forceinline__ unsigned int pk2(float a, float b) {
    unsigned int r;
    asm("v_cvt_pk_bf16_f32 %0, %1, %2" : "=v"(r) : "v"(a), "v"(b));
    return r;
}
__device__ __forceinline__ unsigned short f2bf(float f) {
    unsigned int r;
    asm("v_cvt_pk_bf16_f32 %0, %1, 0" : "=v"(r) : "v"(f));
    return (unsigned short)r;
}
__device__ __forceinline__ uint2 pk4(float4 v) {
    uint2 r;
    r.x = pk2(v.x, v.y);
    r.y = pk2(v.z, v.w);
    return r;
}
__device__ __forceinline__ v8f zero8() {
    v8f r;
#pragma unroll
    for (int i = 0; i < 8; ++i) r[i] = 0.0f;
    return r;
}
// Build a 16-element bf16 fragment from two contiguous 8-half LDS runs (2x ds_load_b128)
__device__ __forceinline__ v16bf mk16(const unsigned short* p0, const unsigned short* p1) {
    v8bf lo = *(const v8bf*)p0;
    v8bf hh = *(const v8bf*)p1;
    v16bf r;
#pragma unroll
    for (int i = 0; i < 8; ++i) { r[i] = lo[i]; r[i + 8] = hh[i]; }
    return r;
}

// ---------- generic NT GEMM: C[M,N] = A[M,K] @ W[N,K]^T (+bias)(+relu)(+resid) ----------
// flags: bit0 = ReLU, bit1 = add residual
// Register-pipelined: global loads of chunk i+1 overlap WMMAs on chunk i.
__global__ __launch_bounds__(128) void gemm_bf16_nt(
    const float* __restrict__ A, const float* __restrict__ Wt,
    const float* __restrict__ bias, const float* __restrict__ resid,
    float* __restrict__ C, int M, int N, int K, int flags)
{
    __shared__ unsigned short As[64 * 32];
    __shared__ unsigned short Ws[64 * 32];
    const int tid  = threadIdx.x;
    const int wave = tid >> 5, lane = tid & 31;
    const int half = lane & 15, hi = lane >> 4;
    const int m0 = blockIdx.y * 64, n0 = blockIdx.x * 64;
    const int nw = wave * 16;

    v8f acc[4];
#pragma unroll
    for (int t = 0; t < 4; ++t) acc[t] = zero8();

    // per-thread staging coordinates (4 float4 slots per matrix)
    int srow[4], scol[4];
#pragma unroll
    for (int j = 0; j < 4; ++j) {
        int f = j * 128 + tid;        // 0..511 float4 slots
        srow[j] = f >> 3;             // 0..63
        scol[j] = (f & 7) * 4;        // 0..28
    }

    float4 ra[4], rw[4];
#pragma unroll
    for (int j = 0; j < 4; ++j) {     // preload chunk 0
        ra[j] = *(const float4*)&A [(size_t)(m0 + srow[j]) * K + scol[j]];
        rw[j] = *(const float4*)&Wt[(size_t)(n0 + srow[j]) * K + scol[j]];
    }

    for (int k0 = 0; k0 < K; k0 += 32) {
        // convert + store current chunk to LDS (packed cvt, single b64 store each)
#pragma unroll
        for (int j = 0; j < 4; ++j) {
            *(uint2*)&As[srow[j] * 32 + scol[j]] = pk4(ra[j]);
            *(uint2*)&Ws[srow[j] * 32 + scol[j]] = pk4(rw[j]);
        }
        __syncthreads();

        // issue global loads for next chunk; latency hidden under the WMMAs below
        if (k0 + 32 < K) {
#pragma unroll
            for (int j = 0; j < 4; ++j) {
                ra[j] = *(const float4*)&A [(size_t)(m0 + srow[j]) * K + (k0 + 32) + scol[j]];
                rw[j] = *(const float4*)&Wt[(size_t)(n0 + srow[j]) * K + (k0 + 32) + scol[j]];
            }
        }

        // B-frag: lane = column (n), K(e) = hi*16 + e (contiguous 16 halfs)
        const int nrow = nw + half;
        v16bf bfrag = mk16(&Ws[nrow * 32 + hi * 16], &Ws[nrow * 32 + hi * 16 + 8]);
#pragma unroll
        for (int mt = 0; mt < 4; ++mt) {
            // A-frag: lane row = M, K(e) = hi*16 + e (contiguous 16 halfs)
            const int arow = mt * 16 + half;
            v16bf afrag = mk16(&As[arow * 32 + hi * 16], &As[arow * 32 + hi * 16 + 8]);
            acc[mt] = __builtin_amdgcn_wmma_f32_16x16x32_bf16(
                false, afrag, false, bfrag, (short)0, acc[mt], false, false);
        }
        __syncthreads();
    }

    // epilogue: C-layout lane holds column n, VGPR v holds row hi*8+v
    const int n  = n0 + nw + half;
    const float bn = bias ? bias[n] : 0.0f;
#pragma unroll
    for (int mt = 0; mt < 4; ++mt) {
#pragma unroll
        for (int v = 0; v < 8; ++v) {
            const int m = m0 + mt * 16 + hi * 8 + v;
            float val = acc[mt][v] + bn;
            if (flags & 1) val = fmaxf(val, 0.0f);
            if (flags & 2) val += resid[(size_t)m * N + n];
            C[(size_t)m * N + n] = val;
        }
    }
}

// ---------- flash attention: one wave per (head, 16-query tile) ----------
__global__ __launch_bounds__(32) void attn_fa(
    const float* __restrict__ qkv, float* __restrict__ O)
{
    __shared__ unsigned short Qs[16 * 64];   // [qrow][feat]
    __shared__ unsigned short Ks[32 * 64];   // [key ][feat]
    __shared__ unsigned short Vt[64 * 32];   // [feat][key]  (transposed)
    __shared__ float          Ss[16 * 32];   // scores
    __shared__ unsigned short Ps[16 * 32];   // probs bf16
    __shared__ float          alph[16];
    __shared__ float          linv[16];

    const int lane = threadIdx.x;
    const int half = lane & 15, hi = lane >> 4;
    const int q0   = blockIdx.x * 16;
    const int h    = blockIdx.y;
    const int qoff = h * DHEAD;
    const int koff = DMODEL + h * DHEAD;
    const int voff = 2 * DMODEL + h * DHEAD;
    const int LDQ  = 3 * DMODEL;

    // stage Q tile (16 x 64) once
#pragma unroll
    for (int i = 0; i < 8; ++i) {
        int f = i * 32 + lane, row = f >> 4, c4 = (f & 15) * 4;
        float4 v = *(const float4*)&qkv[(size_t)(q0 + row) * LDQ + qoff + c4];
        *(uint2*)&Qs[row * 64 + c4] = pk4(v);
    }

    v8f o[4];
#pragma unroll
    for (int t = 0; t < 4; ++t) o[t] = zero8();
    float m_i = -3.0e38f, l_i = 0.0f;

    for (int kb = 0; kb < S_LEN; kb += 32) {
        // stage K chunk (32x64) row-major, V chunk transposed (64x32)
#pragma unroll
        for (int i = 0; i < 16; ++i) {
            int f = i * 32 + lane, row = f >> 4, c4 = (f & 15) * 4;
            float4 vk = *(const float4*)&qkv[(size_t)(kb + row) * LDQ + koff + c4];
            *(uint2*)&Ks[row * 64 + c4] = pk4(vk);
            float4 vv = *(const float4*)&qkv[(size_t)(kb + row) * LDQ + voff + c4];
            Vt[(c4 + 0) * 32 + row] = f2bf(vv.x);
            Vt[(c4 + 1) * 32 + row] = f2bf(vv.y);
            Vt[(c4 + 2) * 32 + row] = f2bf(vv.z);
            Vt[(c4 + 3) * 32 + row] = f2bf(vv.w);
        }
        __syncthreads();

        // S = Q @ K^T : 16x32, contraction over 64 features (2 k-steps) x 2 key tiles
        v16bf aq0 = mk16(&Qs[half * 64 + hi * 16],      &Qs[half * 64 + hi * 16 + 8]);
        v16bf aq1 = mk16(&Qs[half * 64 + 32 + hi * 16], &Qs[half * 64 + 32 + hi * 16 + 8]);
        v8f s0 = zero8(), s1 = zero8();
        {
            v16bf bk;
            bk = mk16(&Ks[half * 64 + hi * 16],            &Ks[half * 64 + hi * 16 + 8]);
            s0 = __builtin_amdgcn_wmma_f32_16x16x32_bf16(false, aq0, false, bk, (short)0, s0, false, false);
            bk = mk16(&Ks[half * 64 + 32 + hi * 16],       &Ks[half * 64 + 32 + hi * 16 + 8]);
            s0 = __builtin_amdgcn_wmma_f32_16x16x32_bf16(false, aq1, false, bk, (short)0, s0, false, false);
            bk = mk16(&Ks[(16 + half) * 64 + hi * 16],     &Ks[(16 + half) * 64 + hi * 16 + 8]);
            s1 = __builtin_amdgcn_wmma_f32_16x16x32_bf16(false, aq0, false, bk, (short)0, s1, false, false);
            bk = mk16(&Ks[(16 + half) * 64 + 32 + hi * 16],&Ks[(16 + half) * 64 + 32 + hi * 16 + 8]);
            s1 = __builtin_amdgcn_wmma_f32_16x16x32_bf16(false, aq1, false, bk, (short)0, s1, false, false);
        }
        // C-layout -> LDS for softmax
#pragma unroll
        for (int v = 0; v < 8; ++v) {
            Ss[(hi * 8 + v) * 32 + half]      = s0[v];
            Ss[(hi * 8 + v) * 32 + 16 + half] = s1[v];
        }
        __syncthreads();

        // online softmax: lanes 0..15 own one query row each
        if (lane < 16) {
            float mx = -3.0e38f;
#pragma unroll 4
            for (int j = 0; j < 32; ++j) mx = fmaxf(mx, Ss[lane * 32 + j]);
            mx *= 0.125f;                         // scale = 1/sqrt(64)
            float mn = fmaxf(m_i, mx);
            float alpha = __expf(m_i - mn);
            float sum = 0.0f;
#pragma unroll 4
            for (int j = 0; j < 32; ++j) {
                float p = __expf(Ss[lane * 32 + j] * 0.125f - mn);
                Ps[lane * 32 + j] = f2bf(p);
                sum += p;
            }
            l_i = l_i * alpha + sum;
            m_i = mn;
            alph[lane] = alpha;
        }
        __syncthreads();

        // rescale O accumulators by per-row alpha
        float ar[8];
#pragma unroll
        for (int v = 0; v < 8; ++v) ar[v] = alph[hi * 8 + v];
#pragma unroll
        for (int t = 0; t < 4; ++t)
#pragma unroll
            for (int v = 0; v < 8; ++v) o[t][v] *= ar[v];

        // O += P(16x32) @ V(32x64) ; V staged transposed so B-frag reads are contiguous
        v16bf ap = mk16(&Ps[half * 32 + hi * 16], &Ps[half * 32 + hi * 16 + 8]);
#pragma unroll
        for (int t = 0; t < 4; ++t) {
            v16bf bv = mk16(&Vt[(t * 16 + half) * 32 + hi * 16],
                            &Vt[(t * 16 + half) * 32 + hi * 16 + 8]);
            o[t] = __builtin_amdgcn_wmma_f32_16x16x32_bf16(false, ap, false, bv, (short)0, o[t], false, false);
        }
        __syncthreads();
    }

    if (lane < 16) linv[lane] = 1.0f / l_i;
    __syncthreads();
    float sr[8];
#pragma unroll
    for (int v = 0; v < 8; ++v) sr[v] = linv[hi * 8 + v];
#pragma unroll
    for (int t = 0; t < 4; ++t)
#pragma unroll
        for (int v = 0; v < 8; ++v)
            O[(size_t)(q0 + hi * 8 + v) * DMODEL + h * DHEAD + t * 16 + half] = o[t][v] * sr[v];
}

// ---------- LayerNorm: one block (256 thr) per row of 512 ----------
__global__ __launch_bounds__(256) void layernorm_k(
    const float* __restrict__ Y, const float* __restrict__ g,
    const float* __restrict__ b, float* __restrict__ X)
{
    __shared__ float red[256];
    const int row = blockIdx.x, tid = threadIdx.x;
    const float x0 = Y[(size_t)row * DMODEL + tid];
    const float x1 = Y[(size_t)row * DMODEL + 256 + tid];
    red[tid] = x0 + x1;
    __syncthreads();
    for (int s = 128; s > 0; s >>= 1) { if (tid < s) red[tid] += red[tid + s]; __syncthreads(); }
    const float mean = red[0] * (1.0f / (float)DMODEL);
    __syncthreads();
    const float d0 = x0 - mean, d1 = x1 - mean;
    red[tid] = d0 * d0 + d1 * d1;
    __syncthreads();
    for (int s = 128; s > 0; s >>= 1) { if (tid < s) red[tid] += red[tid + s]; __syncthreads(); }
    const float rstd = rsqrtf(red[0] * (1.0f / (float)DMODEL) + 1e-5f);
    X[(size_t)row * DMODEL + tid]       = d0 * rstd * g[tid] + b[tid];
    X[(size_t)row * DMODEL + 256 + tid] = d1 * rstd * g[256 + tid] + b[256 + tid];
}

// ---------- patch extraction: (256,256,8) -> (2048, 256) ----------
__global__ __launch_bounds__(256) void patchify_k(
    const float* __restrict__ img, float* __restrict__ P)
{
    const int s = blockIdx.x;           // d*256 + py*16 + px
    const int i = threadIdx.x;          // r*16 + c
    const int d  = s >> 8;
    const int py = (s >> 4) & 15;
    const int px = s & 15;
    const int r = i >> 4, c = i & 15;
    const int hh = py * 16 + r, ww = px * 16 + c;
    P[(size_t)s * 256 + i] = img[((size_t)hh * 256 + ww) * 8 + d];
}

// ---------- sinusoidal positional encoding add ----------
__global__ __launch_bounds__(256) void posenc_k(float* __restrict__ X)
{
    const int s = blockIdx.x, i = threadIdx.x;           // i: pair index 0..255
    const float dv = __expf(-0.0179889460390160f * (float)(2 * i));   // ln(1e4)/512
    const float a  = (float)s * dv;
    X[(size_t)s * DMODEL + 2 * i]     += sinf(a);
    X[(size_t)s * DMODEL + 2 * i + 1] += cosf(a);
}

// ---------- driver ----------
extern "C" void kernel_launch(void* const* d_in, const int* in_sizes, int n_in,
                              void* d_out, int out_size, void* d_ws, size_t ws_size,
                              hipStream_t stream) {
    const float* img   = (const float*)d_in[0];
    const float* W_emb = (const float*)d_in[1];
    const float* b_emb = (const float*)d_in[2];
    const float* Wqkv  = (const float*)d_in[3];
    const float* bqkv  = (const float*)d_in[4];
    const float* Wo    = (const float*)d_in[5];
    const float* bo    = (const float*)d_in[6];
    const float* ln1g  = (const float*)d_in[7];
    const float* ln1b  = (const float*)d_in[8];
    const float* W1    = (const float*)d_in[9];
    const float* b1    = (const float*)d_in[10];
    const float* W2    = (const float*)d_in[11];
    const float* b2    = (const float*)d_in[12];
    const float* ln2g  = (const float*)d_in[13];
    const float* ln2b  = (const float*)d_in[14];
    const float* W_out = (const float*)d_in[15];
    const float* b_out = (const float*)d_in[16];

    float* ws  = (float*)d_ws;
    float* X   = ws;                                // 2048*512
    float* Yb  = X   + (size_t)S_LEN * DMODEL;      // 2048*512
    float* QKV = Yb  + (size_t)S_LEN * DMODEL;      // 2048*1536
    float* Ob  = QKV + (size_t)S_LEN * 3 * DMODEL;  // 2048*512
    float* FF  = Ob  + (size_t)S_LEN * DMODEL;      // 2048*2048
    float* PT  = FF  + (size_t)S_LEN * DFF;         // 2048*256

    dim3 b128(128), b256(256), b32(32);

    patchify_k<<<S_LEN, b256, 0, stream>>>(img, PT);
    gemm_bf16_nt<<<dim3(DMODEL / 64, S_LEN / 64), b128, 0, stream>>>(
        PT, W_emb, b_emb, nullptr, X, S_LEN, DMODEL, 256, 0);
    posenc_k<<<S_LEN, b256, 0, stream>>>(X);

    for (int l = 0; l < NLAYERS; ++l) {
        gemm_bf16_nt<<<dim3(3 * DMODEL / 64, S_LEN / 64), b128, 0, stream>>>(
            X, Wqkv + (size_t)l * 3 * DMODEL * DMODEL, bqkv + l * 3 * DMODEL,
            nullptr, QKV, S_LEN, 3 * DMODEL, DMODEL, 0);
        attn_fa<<<dim3(S_LEN / 16, NHEAD), b32, 0, stream>>>(QKV, Ob);
        gemm_bf16_nt<<<dim3(DMODEL / 64, S_LEN / 64), b128, 0, stream>>>(
            Ob, Wo + (size_t)l * DMODEL * DMODEL, bo + l * DMODEL,
            X, Yb, S_LEN, DMODEL, DMODEL, 2);
        layernorm_k<<<S_LEN, b256, 0, stream>>>(Yb, ln1g + l * DMODEL, ln1b + l * DMODEL, X);
        gemm_bf16_nt<<<dim3(DFF / 64, S_LEN / 64), b128, 0, stream>>>(
            X, W1 + (size_t)l * DFF * DMODEL, b1 + l * DFF,
            nullptr, FF, S_LEN, DFF, DMODEL, 1);
        gemm_bf16_nt<<<dim3(DMODEL / 64, S_LEN / 64), b128, 0, stream>>>(
            FF, W2 + (size_t)l * DMODEL * DFF, b2 + l * DMODEL,
            X, Yb, S_LEN, DMODEL, DFF, 2);
        layernorm_k<<<S_LEN, b256, 0, stream>>>(Yb, ln2g + l * DMODEL, ln2b + l * DMODEL, X);
    }

    gemm_bf16_nt<<<dim3(1536 / 64, S_LEN / 64), b128, 0, stream>>>(
        X, W_out, b_out, nullptr, (float*)d_out, S_LEN, 1536, DMODEL, 0);
}
